// GPT_56985626084084
// MI455X (gfx1250) — compile-verified, matrix operand order checked by reading, not run on hardware
//
#include <hip/hip_runtime.h>
#include <hip/hip_bf16.h>
#include <math.h>

#define H_HEADS 16
#define HD      64
#define DM      1024
#define HMLP    4096
#define VOCAB   32768
#define BQ      2
#define SEQ     2048
#define NTOK    (BQ * SEQ)
#define LGDN    8
#define CHUNK   64
#define NCHUNK  (SEQ / CHUNK)

typedef __attribute__((ext_vector_type(16))) __bf16 v16bf;
typedef __attribute__((ext_vector_type(8)))  __bf16 bf16x8;
typedef __attribute__((ext_vector_type(8)))  float  v8f;
typedef unsigned int u32;
typedef __attribute__((ext_vector_type(4))) u32 v4u;
typedef __attribute__((ext_vector_type(4))) int v4i;
typedef __attribute__((ext_vector_type(8))) int v8i;

#if defined(__has_builtin)
#if __has_builtin(__builtin_amdgcn_tensor_load_to_lds) && \
    __has_builtin(__builtin_amdgcn_s_wait_tensorcnt)
#define USE_TDM 1
#endif
#endif
#ifndef USE_TDM
#define USE_TDM 0
#endif

__device__ __forceinline__ v16bf mk16(bf16x8 lo, bf16x8 hi) {
    v16bf r;
#pragma unroll
    for (int i = 0; i < 8; i++) { r[i] = lo[i]; r[i + 8] = hi[i]; }
    return r;
}

__device__ __forceinline__ v8f wmma_bf16(v16bf a, v16bf b, v8f c) {
    return __builtin_amdgcn_wmma_f32_16x16x32_bf16(false, a, false, b, (short)0, c, false, false);
}

#if USE_TDM
// Issue a TDM 2D tile load: rows lines of BK_ELEMS bf16, line stride = strideElems,
// packed contiguously into LDS at ldsOff. Tracked by TENSORcnt.
__device__ __forceinline__ void tdm_load(const __bf16* g, unsigned ldsOff,
                                         int rows, int tilew, long strideElems) {
    unsigned long long ga = (unsigned long long)g;
    v4u g0;
    g0[0] = 1u;                                        // count=1, user mode
    g0[1] = ldsOff;                                    // lds_addr
    g0[2] = (u32)ga;                                   // global_addr[31:0]
    g0[3] = ((u32)(ga >> 32) & 0x01FFFFFFu) | (2u << 30);  // addr[56:32] | type=2
    v8i g1;
    g1[0] = (int)(1u << 16);                           // mask=0, data_size=1 (2B)
    g1[1] = (int)((u32)tilew << 16);                   // tensor_dim0[15:0] @ [63:48]
    g1[2] = (int)(((u32)rows & 0xFFFFu) << 16);        // tensor_dim1[15:0] @ [95:80]
    g1[3] = (int)(((u32)rows >> 16) | ((u32)tilew << 16));  // dim1 hi | tile_dim0
    g1[4] = (int)((u32)rows & 0xFFFFu);                // tile_dim1 | tile_dim2=0
    unsigned long long sb = (unsigned long long)strideElems;
    g1[5] = (int)(u32)sb;                              // tensor_dim0_stride[31:0]
    g1[6] = (int)((u32)(sb >> 32) & 0xFFFFu);          // stride[47:32] | dim1_stride lo
    g1[7] = 0;
    v4i gz; gz[0] = 0; gz[1] = 0; gz[2] = 0; gz[3] = 0;
#if __clang_major__ >= 23
    v8i gz8;
#pragma unroll
    for (int i = 0; i < 8; i++) gz8[i] = 0;
    __builtin_amdgcn_tensor_load_to_lds(g0, g1, gz, gz, gz8, 0);
#else
    __builtin_amdgcn_tensor_load_to_lds(g0, g1, gz, gz, 0);
#endif
}
#endif

// ---------------------------------------------------------------- fp32 -> bf16 conversion
__global__ __launch_bounds__(256) void k_cvt(const float* __restrict__ in,
                                             __bf16* __restrict__ out, long n) {
    long i = ((long)blockIdx.x * 256 + threadIdx.x) * 4;
    if (i + 3 < n) {
        float4 v = *(const float4*)(in + i);
        out[i + 0] = (__bf16)v.x;
        out[i + 1] = (__bf16)v.y;
        out[i + 2] = (__bf16)v.z;
        out[i + 3] = (__bf16)v.w;
    }
}

// ---------------------------------------------------------------- embedding
__global__ void k_embed(const int* __restrict__ ids, const float* __restrict__ emb,
                        float* __restrict__ x) {
    int t = blockIdx.x;
    const float* src = emb + (long)ids[t] * DM;
    float* dst = x + (long)t * DM;
    for (int i = threadIdx.x; i < DM; i += blockDim.x) dst[i] = src[i];
}

// ---------------------------------------------------------------- rmsnorm (fp32 in, bf16 out)
__global__ __launch_bounds__(256) void k_rmsnorm(const float* __restrict__ x,
                                                 const float* __restrict__ w,
                                                 __bf16* __restrict__ y, float eps) {
    __shared__ float red[8];
    int tk = blockIdx.x;
    const float* xr = x + (long)tk * DM;
    float ss = 0.f;
    for (int i = threadIdx.x; i < DM; i += 256) { float v = xr[i]; ss += v * v; }
#pragma unroll
    for (int m = 1; m < 32; m <<= 1) ss += __shfl_xor(ss, m, 32);
    if ((threadIdx.x & 31) == 0) red[threadIdx.x >> 5] = ss;
    __syncthreads();
    if (threadIdx.x == 0) {
        float t = 0.f;
        for (int i = 0; i < 8; i++) t += red[i];
        red[0] = t;
    }
    __syncthreads();
    float scale = rsqrtf(red[0] / (float)DM + eps);
    __bf16* yr = y + (long)tk * DM;
    for (int i = threadIdx.x; i < DM; i += 256) yr[i] = (__bf16)(xr[i] * scale * w[i]);
}

// ---------------------------------------------------------------- bf16 WMMA GEMM
// C[M,N](f32, optional) / Cb[M,N](bf16, optional) = resid + alpha*act(A[M,K]*W[N,K]^T)
// A,W are bf16. 128x256 block tile, 64x64 wave tile, double-buffered LDS.
// Staging: TDM (tensor_load_to_lds, TENSORcnt) when available, else uint4 copies.
#define BM 128
#define BN 256
#define BK 32

__global__ __launch_bounds__(256) void k_gemm(const __bf16* __restrict__ A, int lda,
                                              const __bf16* __restrict__ W, int ldw,
                                              float* __restrict__ C,
                                              __bf16* __restrict__ Cb, int ldc,
                                              const float* __restrict__ resid,
                                              float alpha, int K, int act) {
    __shared__ __bf16 As[2][BM][BK];   // 16 KB
    __shared__ __bf16 Bs[2][BN][BK];   // 32 KB

    const int m0 = blockIdx.y * BM, n0 = blockIdx.x * BN;
    const int tid  = threadIdx.x;
    const int lane = tid & 31, wv = tid >> 5;
    const int wm = wv >> 2, wn = wv & 3;        // 2 (M) x 4 (N) wave grid
    const int l15 = lane & 15, hi = lane >> 4;
    const int nk = K / BK;

#if USE_TDM
    unsigned aOff = (unsigned)(unsigned long long)(void*)&As[0][0][0];
    unsigned bOff = (unsigned)(unsigned long long)(void*)&Bs[0][0][0];
    auto stage = [&](int buf, int k0) {
        if (wv == 0) {
            tdm_load(A + (long)m0 * lda + k0, aOff + buf * (BM * BK * 2), BM, BK, lda);
            tdm_load(W + (long)n0 * ldw + k0, bOff + buf * (BN * BK * 2), BN, BK, ldw);
        }
    };
    auto stage_wait = [&]() {
        if (wv == 0) __builtin_amdgcn_s_wait_tensorcnt(0);
    };
#else
    auto stage = [&](int buf, int k0) {
#pragma unroll
        for (int p = 0; p < 2; p++) {            // A: 128x32 bf16 = 512 uint4
            int idx = p * 256 + tid;
            int r = idx >> 2, c = (idx & 3) * 8;
            *(uint4*)&As[buf][r][c] = *(const uint4*)(A + (long)(m0 + r) * lda + k0 + c);
        }
#pragma unroll
        for (int p = 0; p < 4; p++) {            // B: 256x32 bf16 = 1024 uint4
            int idx = p * 256 + tid;
            int r = idx >> 2, c = (idx & 3) * 8;
            *(uint4*)&Bs[buf][r][c] = *(const uint4*)(W + (long)(n0 + r) * ldw + k0 + c);
        }
    };
    auto stage_wait = [&]() {};
#endif

    v8f acc[4][4];
#pragma unroll
    for (int mi = 0; mi < 4; mi++)
#pragma unroll
        for (int ni = 0; ni < 4; ni++)
#pragma unroll
            for (int r = 0; r < 8; r++) acc[mi][ni][r] = 0.f;

    stage(0, 0);
    stage_wait();
    __syncthreads();

    for (int kt = 0; kt < nk; kt++) {
        int cur = kt & 1;
        if (kt + 1 < nk) stage(cur ^ 1, (kt + 1) * BK);
        if (kt + 2 < nk) {                       // L2 prefetch two tiles ahead
            __builtin_prefetch(W + (long)(n0 + tid) * ldw + (kt + 2) * BK, 0, 1);
            if (tid < BM)
                __builtin_prefetch(A + (long)(m0 + tid) * lda + (kt + 2) * BK, 0, 1);
        }

        v16bf af[4], bfr[4];
#pragma unroll
        for (int mi = 0; mi < 4; mi++) {
            int row = wm * 64 + mi * 16 + l15;
            af[mi] = mk16(*(const bf16x8*)&As[cur][row][hi * 8],
                          *(const bf16x8*)&As[cur][row][16 + hi * 8]);
        }
#pragma unroll
        for (int ni = 0; ni < 4; ni++) {
            int row = wn * 64 + ni * 16 + l15;
            bfr[ni] = mk16(*(const bf16x8*)&Bs[cur][row][hi * 8],
                           *(const bf16x8*)&Bs[cur][row][16 + hi * 8]);
        }
#pragma unroll
        for (int mi = 0; mi < 4; mi++)
#pragma unroll
            for (int ni = 0; ni < 4; ni++)
                acc[mi][ni] = wmma_bf16(af[mi], bfr[ni], acc[mi][ni]);

        stage_wait();
        __syncthreads();
    }

#pragma unroll
    for (int mi = 0; mi < 4; mi++)
#pragma unroll
        for (int ni = 0; ni < 4; ni++)
#pragma unroll
            for (int r = 0; r < 8; r++) {
                long row = m0 + wm * 64 + mi * 16 + hi * 8 + r;
                long col = n0 + wn * 64 + ni * 16 + l15;
                float v = acc[mi][ni][r];
                if (act) v = v / (1.f + expf(-v));   // SiLU
                v *= alpha;
                if (resid) v += resid[row * (long)ldc + col];
                if (C)  C[row * (long)ldc + col] = v;
                if (Cb) Cb[row * (long)ldc + col] = (__bf16)v;
            }
}

// ---------------------------------------------------------------- GDN elementwise gate
__global__ __launch_bounds__(32) void k_gdn_gate(float* __restrict__ proj) {
    int g = blockIdx.x;                 // t*16 + h
    int t = g >> 4, h = g & 15;
    int lane = threadIdx.x;
    long base = (long)t * 5 * DM + h * HD;

    float q0 = proj[base + lane], q1 = proj[base + lane + 32];
    float ss = q0 * q0 + q1 * q1;
#pragma unroll
    for (int m = 1; m < 32; m <<= 1) ss += __shfl_xor(ss, m, 32);
    float sc = rsqrtf(ss / 64.f + 1.1920929e-07f);
    proj[base + lane] = q0 * sc;
    proj[base + lane + 32] = q1 * sc;

#pragma unroll
    for (int j = 0; j < 2; j++) {
        int d = lane + j * 32;
        float kk = tanhf(proj[base + DM + d]);
        float vv = proj[base + 2 * DM + d];
        float aa = proj[base + 3 * DM + d];
        aa = 1.f / (1.f + expf(-(aa + 2.f)));
        aa = fminf(fmaxf(aa, 0.6f), 0.9995f);
        proj[base + 3 * DM + d] = aa;
        proj[base + DM + d] = kk * vv;   // u
    }
}

// ---------------------------------------------------------------- GDN scan phase 1 (chunk-parallel)
__global__ __launch_bounds__(64) void k_scan1(const float* __restrict__ proj,
                                              float* __restrict__ s0,
                                              float* __restrict__ pfx) {
    int blk = blockIdx.x;                // ((b*16 + h) * NCHUNK) + c
    int c = blk & (NCHUNK - 1);
    int bh = blk >> 5;
    int h = bh & 15, b = bh >> 4;
    int d = threadIdx.x;
    float prefix = 1.f, acc = 0.f;
    for (int p = 0; p < CHUNK; p++) {
        long t = (long)b * SEQ + c * CHUNK + p;
        long base = t * 5 * DM + h * HD + d;
        float a = proj[base + 3 * DM];
        float u = proj[base + DM];
        prefix *= a;
        acc += u / fmaxf(prefix, 1e-6f);
        long o = t * DM + h * HD + d;
        s0[o]  = prefix * acc;
        pfx[o] = prefix;
    }
}

// ---------------------------------------------------------------- GDN scan phase 2 (cross-chunk carry)
__global__ __launch_bounds__(64) void k_scan2(const float* __restrict__ s0,
                                              const float* __restrict__ pfx,
                                              float* __restrict__ carry) {
    int bh = blockIdx.x;
    int h = bh & 15, b = bh >> 4;
    int d = threadIdx.x;
    float cy = 0.f;
    for (int c = 0; c < NCHUNK; c++) {
        carry[((long)bh * NCHUNK + c) * HD + d] = cy;
        long t = (long)b * SEQ + c * CHUNK + CHUNK - 1;
        long o = t * DM + h * HD + d;
        cy = pfx[o] * cy + s0[o];
    }
}

// ---------------------------------------------------------------- GDN output mix (bf16 out)
__global__ __launch_bounds__(256) void k_gdn_out(const float* __restrict__ proj,
                                                 const float* __restrict__ s0,
                                                 const float* __restrict__ pfx,
                                                 const float* __restrict__ carry,
                                                 __bf16* __restrict__ y) {
    long i = (long)blockIdx.x * 256 + threadIdx.x;     // over NTOK*DM
    long t = i / DM, dd = i % DM;
    int h = (int)(dd >> 6), d = (int)(dd & 63);
    int b = (int)(t / SEQ), s = (int)(t % SEQ), c = s >> 6;
    long base = t * 5 * DM + dd;
    float q = proj[base];
    float v = proj[base + 2 * DM];
    float g = proj[base + 4 * DM];
    float cy = carry[(((long)(b * 16 + h)) * NCHUNK + c) * HD + d];
    float state = pfx[i] * cy + s0[i];
    float gate = 1.f / (1.f + expf(-g));
    y[i] = (__bf16)(gate * (q * state) + (1.f - gate) * v);
}

// ---------------------------------------------------------------- flash attention (bf16 qkv, 1 wave / q-tile)
__global__ __launch_bounds__(32) void k_attn(const __bf16* __restrict__ qkv,
                                             __bf16* __restrict__ y) {
    int qt = blockIdx.x, h = blockIdx.y, b = blockIdx.z;
    int lane = threadIdx.x, l15 = lane & 15, hi = lane >> 4;
    __shared__ float pl[16][16];

    v16bf qa[2];
    {
        const __bf16* qr = qkv + (long)(b * SEQ + qt * 16 + l15) * 3 * DM + h * HD;
#pragma unroll
        for (int f = 0; f < 2; f++)
            qa[f] = mk16(*(const bf16x8*)(qr + f * 32 + hi * 8),
                         *(const bf16x8*)(qr + f * 32 + 16 + hi * 8));
    }

    v8f yacc[4];
    float mrun[8], srun[8];
#pragma unroll
    for (int nt = 0; nt < 4; nt++)
#pragma unroll
        for (int r = 0; r < 8; r++) yacc[nt][r] = 0.f;
#pragma unroll
    for (int r = 0; r < 8; r++) { mrun[r] = -1e30f; srun[r] = 0.f; }

    for (int kb = 0; kb <= qt; kb++) {
        v16bf kf[2];
        {
            const __bf16* kr = qkv + (long)(b * SEQ + kb * 16 + l15) * 3 * DM + DM + h * HD;
#pragma unroll
            for (int f = 0; f < 2; f++)
                kf[f] = mk16(*(const bf16x8*)(kr + f * 32 + hi * 8),
                             *(const bf16x8*)(kr + f * 32 + 16 + hi * 8));
        }
        v8f s;
#pragma unroll
        for (int r = 0; r < 8; r++) s[r] = 0.f;
        s = wmma_bf16(qa[0], kf[0], s);
        s = wmma_bf16(qa[1], kf[1], s);
#pragma unroll
        for (int r = 0; r < 8; r++) s[r] *= 0.125f;   // 1/sqrt(hd)

        if (kb == qt) {
#pragma unroll
            for (int r = 0; r < 8; r++)
                if (l15 > hi * 8 + r) s[r] = -1e30f;
        }
        float scl[8];
#pragma unroll
        for (int r = 0; r < 8; r++) {
            float v = s[r];
#pragma unroll
            for (int m = 1; m < 16; m <<= 1) v = fmaxf(v, __shfl_xor(v, m, 32));
            float nm = fmaxf(mrun[r], v);
            scl[r] = expf(mrun[r] - nm);
            mrun[r] = nm;
        }
#pragma unroll
        for (int r = 0; r < 8; r++) {
            s[r] = expf(s[r] - mrun[r]);
            float v = s[r];
#pragma unroll
            for (int m = 1; m < 16; m <<= 1) v += __shfl_xor(v, m, 32);
            srun[r] = srun[r] * scl[r] + v;
        }
#pragma unroll
        for (int nt = 0; nt < 4; nt++)
#pragma unroll
            for (int r = 0; r < 8; r++) yacc[nt][r] *= scl[r];

        __syncthreads();
#pragma unroll
        for (int r = 0; r < 8; r++) pl[hi * 8 + r][l15] = s[r];
        __syncthreads();
        v16bf pa;
#pragma unroll
        for (int j = 0; j < 8; j++) {
            pa[j] = (__bf16)pl[l15][hi * 8 + j];
            pa[j + 8] = (__bf16)0.f;
        }
#pragma unroll
        for (int nt = 0; nt < 4; nt++) {
            v16bf vb;
#pragma unroll
            for (int j = 0; j < 8; j++) {
                int key = hi * 8 + j;
                vb[j] = qkv[(long)(b * SEQ + kb * 16 + key) * 3 * DM +
                            2 * DM + h * HD + nt * 16 + l15];
                vb[j + 8] = (__bf16)0.f;
            }
            yacc[nt] = wmma_bf16(pa, vb, yacc[nt]);
        }
    }
#pragma unroll
    for (int nt = 0; nt < 4; nt++)
#pragma unroll
        for (int r = 0; r < 8; r++) {
            long row = (long)b * SEQ + qt * 16 + hi * 8 + r;
            y[row * DM + h * HD + nt * 16 + l15] = (__bf16)(yacc[nt][r] / srun[r]);
        }
}

// ---------------------------------------------------------------- fused LM head + NLL
// 128 tokens/block, bf16 x-tile in 256KB dynamic LDS; one emb B-fragment feeds
// 8 WMMAs; online logsumexp in wave-private LDS slots.
#define LTM 128

__global__ __launch_bounds__(256) void k_loss(const __bf16* __restrict__ xn,
                                              const __bf16* __restrict__ emb,
                                              const float* __restrict__ bias,
                                              const int* __restrict__ tgt,
                                              float* __restrict__ nll) {
    extern __shared__ unsigned char smem[];
    __bf16* xt   = (__bf16*)smem;                              // LTM*DM bf16
    float*  wm   = (float*)(smem + (long)LTM * DM * 2);        // 8*LTM
    float*  wsv  = wm + 8 * LTM;                               // 8*LTM
    float*  tlog = wsv + 8 * LTM;                              // LTM
    int*    tg   = (int*)(tlog + LTM);                         // LTM

    int tile = blockIdx.x;
    int tid = threadIdx.x;
    const uint4* src = (const uint4*)(xn + (long)tile * LTM * DM);
    for (int i = tid; i < LTM * DM / 8; i += 256) ((uint4*)xt)[i] = src[i];
    for (int i = tid; i < 8 * LTM; i += 256) { wm[i] = -1e30f; wsv[i] = 0.f; }
    if (tid < LTM) { tg[tid] = tgt[tile * LTM + tid]; tlog[tid] = 0.f; }
    __syncthreads();

    int lane = tid & 31, wv = tid >> 5, l15 = lane & 15, hi = lane >> 4;

    for (int nt = wv; nt < VOCAB / 16; nt += 8) {
        int n0 = nt * 16;
        v8f acc[8];
#pragma unroll
        for (int mf = 0; mf < 8; mf++)
#pragma unroll
            for (int r = 0; r < 8; r++) acc[mf][r] = 0.f;

        const __bf16* er = emb + (long)(n0 + l15) * DM;
        for (int k0 = 0; k0 < DM; k0 += 32) {
            v16bf bfr = mk16(*(const bf16x8*)(er + k0 + hi * 8),
                             *(const bf16x8*)(er + k0 + 16 + hi * 8));
#pragma unroll
            for (int mf = 0; mf < 8; mf++) {
                const __bf16* xr = xt + (long)(mf * 16 + l15) * DM + k0;
                v16bf af = mk16(*(const bf16x8*)(xr + hi * 8),
                                *(const bf16x8*)(xr + 16 + hi * 8));
                acc[mf] = wmma_bf16(af, bfr, acc[mf]);
            }
        }

        float bs = bias[n0 + l15];
#pragma unroll
        for (int mf = 0; mf < 8; mf++) {
#pragma unroll
            for (int r = 0; r < 8; r++) acc[mf][r] += bs;
#pragma unroll
            for (int r = 0; r < 8; r++) {
                int row = mf * 16 + hi * 8 + r;
                int t = tg[row];
                if (t >= n0 && t < n0 + 16 && l15 == (t - n0)) tlog[row] = acc[mf][r];
            }
            float rmax[8], rsum[8];
#pragma unroll
            for (int r = 0; r < 8; r++) {
                float v = acc[mf][r];
#pragma unroll
                for (int m = 1; m < 16; m <<= 1) v = fmaxf(v, __shfl_xor(v, m, 32));
                rmax[r] = v;
            }
#pragma unroll
            for (int r = 0; r < 8; r++) {
                float e = expf(acc[mf][r] - rmax[r]);
#pragma unroll
                for (int m = 1; m < 16; m <<= 1) e += __shfl_xor(e, m, 32);
                rsum[r] = e;
            }
            if (l15 == 0) {                     // wave-private stats, lanes 0 & 16
#pragma unroll
                for (int r = 0; r < 8; r++) {
                    int row = mf * 16 + hi * 8 + r;
                    float om = wm[wv * LTM + row];
                    float nm = fmaxf(om, rmax[r]);
                    wsv[wv * LTM + row] = wsv[wv * LTM + row] * expf(om - nm) +
                                          rsum[r] * expf(rmax[r] - nm);
                    wm[wv * LTM + row] = nm;
                }
            }
        }
    }
    __syncthreads();
    if (tid < LTM) {
        float M = -1e30f;
        for (int w = 0; w < 8; w++) M = fmaxf(M, wm[w * LTM + tid]);
        float S = 0.f;
        for (int w = 0; w < 8; w++) S += wsv[w * LTM + tid] * expf(wm[w * LTM + tid] - M);
        nll[tile * LTM + tid] = (M + logf(S)) - tlog[tid];
    }
}

__global__ __launch_bounds__(256) void k_mean(const float* __restrict__ nll,
                                              float* __restrict__ out) {
    __shared__ float red[8];
    float s = 0.f;
    for (int i = threadIdx.x; i < NTOK; i += 256) s += nll[i];
#pragma unroll
    for (int m = 1; m < 32; m <<= 1) s += __shfl_xor(s, m, 32);
    if ((threadIdx.x & 31) == 0) red[threadIdx.x >> 5] = s;
    __syncthreads();
    if (threadIdx.x == 0) {
        float t = 0.f;
        for (int i = 0; i < 8; i++) t += red[i];
        out[0] = t / (float)NTOK;
    }
}

// ---------------------------------------------------------------- host side
static void gemm(const __bf16* A, int lda, const __bf16* W, int ldw,
                 float* C, __bf16* Cb, int ldc, const float* resid, float alpha,
                 int M, int N, int K, int act, hipStream_t st) {
    dim3 g(N / BN, M / BM);
    k_gemm<<<g, 256, 0, st>>>(A, lda, W, ldw, C, Cb, ldc, resid, alpha, K, act);
}

static void cvt(const float* in, __bf16* out, long n, hipStream_t st) {
    k_cvt<<<(unsigned)(n / 1024), 256, 0, st>>>(in, out, n);
}

extern "C" void kernel_launch(void* const* d_in, const int* in_sizes, int n_in,
                              void* d_out, int out_size, void* d_ws, size_t ws_size,
                              hipStream_t stream) {
    const int*   ids  = (const int*)d_in[0];
    const int*   tgt  = (const int*)d_in[1];
    const float* emb  = (const float*)d_in[2];
    const float* lmb  = (const float*)d_in[3];
    const float* gin  = (const float*)d_in[4];   // [8, 5D, D]
    const float* gout = (const float*)d_in[5];   // [8, D, D]
    const float* gn1  = (const float*)d_in[6];
    const float* gn2  = (const float*)d_in[7];
    const float* gfc  = (const float*)d_in[8];   // [8, Hm, D]
    const float* gpj  = (const float*)d_in[9];   // [8, D, Hm]
    const float* aqkv = (const float*)d_in[10];
    const float* apj  = (const float*)d_in[11];
    const float* an1  = (const float*)d_in[12];
    const float* an2  = (const float*)d_in[13];
    const float* afc  = (const float*)d_in[14];
    const float* ampj = (const float*)d_in[15];
    const float* fnw  = (const float*)d_in[16];

    // fp32 scratch
    float* ws    = (float*)d_ws;
    float* x     = ws;
    float* proj  = x    + (long)NTOK * DM;
    float* s0b   = proj + (long)NTOK * 5 * DM;
    float* pfxb  = s0b  + (long)NTOK * DM;
    float* nllb  = pfxb + (long)NTOK * DM;
    float* carry = nllb + NTOK;                  // B*H*NCHUNK*HD
    // bf16 scratch
    __bf16* bp    = (__bf16*)(carry + (long)BQ * H_HEADS * NCHUNK * HD);
    __bf16* xnb   = bp;     bp += (long)NTOK * DM;
    __bf16* ybb   = bp;     bp += (long)NTOK * DM;
    __bf16* hbb   = bp;     bp += (long)NTOK * HMLP;
    __bf16* qkvb  = bp;     bp += (long)NTOK * 3 * DM;
    __bf16* ginb  = bp;     bp += (long)LGDN * 5 * DM * DM;
    __bf16* goutb = bp;     bp += (long)LGDN * DM * DM;
    __bf16* gfcb  = bp;     bp += (long)LGDN * HMLP * DM;
    __bf16* gpjb  = bp;     bp += (long)LGDN * DM * HMLP;
    __bf16* aqkvb = bp;     bp += (long)3 * DM * DM;
    __bf16* apjb  = bp;     bp += (long)DM * DM;
    __bf16* afcb  = bp;     bp += (long)HMLP * DM;
    __bf16* ampjb = bp;     bp += (long)DM * HMLP;
    __bf16* embb  = bp;     bp += (long)VOCAB * DM;

    // one-time weight conversion (per launch; deterministic)
    cvt(gin,  ginb,  (long)LGDN * 5 * DM * DM, stream);
    cvt(gout, goutb, (long)LGDN * DM * DM, stream);
    cvt(gfc,  gfcb,  (long)LGDN * HMLP * DM, stream);
    cvt(gpj,  gpjb,  (long)LGDN * DM * HMLP, stream);
    cvt(aqkv, aqkvb, (long)3 * DM * DM, stream);
    cvt(apj,  apjb,  (long)DM * DM, stream);
    cvt(afc,  afcb,  (long)HMLP * DM, stream);
    cvt(ampj, ampjb, (long)DM * HMLP, stream);
    cvt(emb,  embb,  (long)VOCAB * DM, stream);

    k_embed<<<NTOK, 256, 0, stream>>>(ids, emb, x);

    for (int i = 0; i < LGDN; i++) {
        float rsc = 1.f / sqrtf(2.f * (float)(i + 1));
        k_rmsnorm<<<NTOK, 256, 0, stream>>>(x, gn1 + (long)i * DM, xnb, 1e-6f);
        gemm(xnb, DM, ginb + (long)i * 5 * DM * DM, DM, proj, nullptr, 5 * DM,
             nullptr, 1.f, NTOK, 5 * DM, DM, 0, stream);
        k_gdn_gate<<<NTOK * H_HEADS, 32, 0, stream>>>(proj);
        k_scan1<<<BQ * H_HEADS * NCHUNK, 64, 0, stream>>>(proj, s0b, pfxb);
        k_scan2<<<BQ * H_HEADS, 64, 0, stream>>>(s0b, pfxb, carry);
        k_gdn_out<<<NTOK * DM / 256, 256, 0, stream>>>(proj, s0b, pfxb, carry, ybb);
        gemm(ybb, DM, goutb + (long)i * DM * DM, DM, x, nullptr, DM, x, rsc,
             NTOK, DM, DM, 0, stream);
        k_rmsnorm<<<NTOK, 256, 0, stream>>>(x, gn2 + (long)i * DM, xnb, 1e-6f);
        gemm(xnb, DM, gfcb + (long)i * HMLP * DM, DM, nullptr, hbb, HMLP,
             nullptr, 1.f, NTOK, HMLP, DM, 1 /*silu*/, stream);
        gemm(hbb, HMLP, gpjb + (long)i * DM * HMLP, HMLP, x, nullptr, DM, x, rsc,
             NTOK, DM, HMLP, 0, stream);
    }

    float rsc = 1.f / sqrtf(2.f * (float)(LGDN + 1));
    k_rmsnorm<<<NTOK, 256, 0, stream>>>(x, an1, xnb, 1e-6f);
    gemm(xnb, DM, aqkvb, DM, nullptr, qkvb, 3 * DM, nullptr, 1.f,
         NTOK, 3 * DM, DM, 0, stream);
    k_attn<<<dim3(SEQ / 16, H_HEADS, BQ), 32, 0, stream>>>(qkvb, ybb);
    gemm(ybb, DM, apjb, DM, x, nullptr, DM, x, rsc, NTOK, DM, DM, 0, stream);
    k_rmsnorm<<<NTOK, 256, 0, stream>>>(x, an2, xnb, 1e-6f);
    gemm(xnb, DM, afcb, DM, nullptr, hbb, HMLP, nullptr, 1.f,
         NTOK, HMLP, DM, 1, stream);
    gemm(hbb, HMLP, ampjb, HMLP, x, nullptr, DM, x, rsc, NTOK, DM, HMLP, 0, stream);

    k_rmsnorm<<<NTOK, 256, 0, stream>>>(x, fnw, xnb, 1e-6f);
    size_t loss_smem = (size_t)LTM * DM * 2 + (16 * LTM + 2 * LTM) * 4 + 64;
    k_loss<<<NTOK / LTM, 256, loss_smem, stream>>>(xnb, embb, lmb, tgt, nllb);
    k_mean<<<1, 256, 0, stream>>>(nllb, (float*)d_out);
}
